// LSTMAutoEncoder_48034914238758
// MI455X (gfx1250) — matrix operand order repaired and checked by reading, not compiled, not run on hardware
//
#include <hip/hip_runtime.h>
#include <hip/hip_bf16.h>

#define BB 128
#define TT 256
#define FF 128
#define HH 768

typedef __attribute__((ext_vector_type(16))) _Float16 v16h;
typedef __attribute__((ext_vector_type(8)))  _Float16 v8h;
typedef __attribute__((ext_vector_type(8)))  float    v8f;

// ---------------------------------------------------------------------------
// Fragment loaders (wave32); A and B both use contiguous k = 16*half + s order.
// ---------------------------------------------------------------------------

// A matrix (16x32 f16, MxK), row-major source with row stride `ld` elements.
// Lanes 0-15 => M=0..15, K kc+0..15; lanes 16-31 => same M,
// K kc+16..31. Two contiguous 16B loads per lane.
__device__ __forceinline__ v16h load_a_frag(const _Float16* __restrict__ base,
                                            int ld, int m0, int kc, int lane) {
  const int half = lane >> 4;
  const _Float16* p = base + (size_t)(m0 + (lane & 15)) * ld + kc + 16 * half;
  v8h lo = *(const v8h*)p;         // K = kc + 16*half + [0..8)
  v8h hi = *(const v8h*)(p + 8);   // K = kc + 16*half + 8 + [0..8)
  return __builtin_shufflevector(lo, hi, 0, 1, 2, 3, 4, 5, 6, 7,
                                 8, 9, 10, 11, 12, 13, 14, 15);
}

// B matrix (32x16 f16, KxN) with B[k][n] = W[n][k], W row-major [N][ldk].
// N = lane&15; lanes 0-15 hold K = kc+0..15, lanes 16-31 hold K = kc+16..31
// -> one contiguous 32-byte load straight from the weight row.
__device__ __forceinline__ v16h load_b_frag(const _Float16* __restrict__ W,
                                            int ldk, int n0, int kc, int lane) {
  const _Float16* p = W + (size_t)(n0 + (lane & 15)) * ldk + kc + 16 * (lane >> 4);
  return *(const v16h*)p;
}

__device__ __forceinline__ float sigmoid_f(float x) {
  return 1.0f / (1.0f + __expf(-x));
}

// One LSTM cell-step descriptor (POD so it can be a by-value kernel arg).
struct CellArgs {
  const _Float16* xin;   // [BB][Kx] with row stride ldx
  int ldx;
  int Kx;
  const _Float16* Wih;   // [4*HH][Kx] row-major
  const _Float16* hin;   // [BB][HH]
  const _Float16* Whh;   // [4*HH][HH] row-major
  const float* bias;     // [4*HH]
  float* c;              // [BB][HH] fp32 state (read-modify-write)
  _Float16* hout;        // [BB][HH]
};

// ---------------------------------------------------------------------------
// Fused LSTM cell step:
//   gates = x @ Wih.T + h @ Whh.T + b ; c' = sig(f)*c + sig(i)*tanh(g)
//   h'    = sig(o)*tanh(c')
// Grid: (HH/16, BB/32, ncells), 64 threads = 2 waves. Small blocks spread
// the 384 waves/cell across many WGPs (no inter-wave sharing -> free).
// blockIdx.z selects which of the two independent cell steps to run, so the
// encoder can run layer1(t-1) and layer0(t) in a single launch.
// ---------------------------------------------------------------------------
__global__ __launch_bounds__(64) void lstm_cell_kernel(CellArgs a0, CellArgs a1) {
  const CellArgs p = (blockIdx.z == 0) ? a0 : a1;

  const int lane = threadIdx.x & 31;
  const int wave = threadIdx.x >> 5;
  const int m0 = blockIdx.y * 32 + wave * 16;  // batch-row tile
  const int jb = blockIdx.x * 16;              // column slab within H

  v8f acc[4];                                  // i, f, g, o accumulators (fp32)
#pragma unroll
  for (int g = 0; g < 4; ++g) acc[g] = v8f{};

  // Phase 1: x @ Wih.T contribution.
  for (int kc = 0; kc < p.Kx; kc += 32) {
    v16h a = load_a_frag(p.xin, p.ldx, m0, kc, lane);
#pragma unroll
    for (int g = 0; g < 4; ++g) {
      v16h b = load_b_frag(p.Wih, p.Kx, g * HH + jb, kc, lane);
      acc[g] = __builtin_amdgcn_wmma_f32_16x16x32_f16(
          false, a, false, b, (short)0, acc[g], false, false);
    }
  }

  // Phase 2: h @ Whh.T contribution (K = HH).
  for (int kc = 0; kc < HH; kc += 32) {
    v16h a = load_a_frag(p.hin, HH, m0, kc, lane);
#pragma unroll
    for (int g = 0; g < 4; ++g) {
      v16h b = load_b_frag(p.Whh, HH, g * HH + jb, kc, lane);
      acc[g] = __builtin_amdgcn_wmma_f32_16x16x32_f16(
          false, a, false, b, (short)0, acc[g], false, false);
    }
  }

  // Epilogue: D layout => element r: m = m0 + r + 8*(lane>>4), n = lane&15.
  const int half = lane >> 4;
  const int j = jb + (lane & 15);
  const float bi = p.bias[0 * HH + j];
  const float bf = p.bias[1 * HH + j];
  const float bg = p.bias[2 * HH + j];
  const float bo = p.bias[3 * HH + j];
#pragma unroll
  for (int r = 0; r < 8; ++r) {
    const int m = m0 + r + 8 * half;
    const float si = sigmoid_f(acc[0][r] + bi);
    const float sf = sigmoid_f(acc[1][r] + bf);
    const float tg = tanhf(acc[2][r] + bg);
    const float so = sigmoid_f(acc[3][r] + bo);
    const size_t idx = (size_t)m * HH + j;
    const float c2 = sf * p.c[idx] + si * tg;
    p.c[idx] = c2;
    p.hout[idx] = (_Float16)(so * tanhf(c2));
  }
}

// ---------------------------------------------------------------------------
// Output projection: y = h @ lin_W.T + lin_b, fp32 into out[:, tout, :]
// (time-flipped slot) and f16 feedback into xd for the next decoder step.
// Grid: (FF/16, BB/32), 64 threads.
// ---------------------------------------------------------------------------
__global__ __launch_bounds__(64) void linear_out_kernel(
    const _Float16* __restrict__ h,
    const _Float16* __restrict__ W,
    const float* __restrict__ b,
    float* __restrict__ out,
    _Float16* __restrict__ xd,
    int tout) {
  const int lane = threadIdx.x & 31;
  const int wave = threadIdx.x >> 5;
  const int m0 = blockIdx.y * 32 + wave * 16;
  const int n0 = blockIdx.x * 16;

  v8f acc = v8f{};
  for (int kc = 0; kc < HH; kc += 32) {
    v16h a = load_a_frag(h, HH, m0, kc, lane);
    v16h bf = load_b_frag(W, HH, n0, kc, lane);
    acc = __builtin_amdgcn_wmma_f32_16x16x32_f16(
        false, a, false, bf, (short)0, acc, false, false);
  }

  const int half = lane >> 4;
  const int n = n0 + (lane & 15);
  const float bn = b[n];
#pragma unroll
  for (int r = 0; r < 8; ++r) {
    const int m = m0 + r + 8 * half;
    const float y = acc[r] + bn;
    out[((size_t)m * TT + tout) * FF + n] = y;
    xd[(size_t)m * FF + n] = (_Float16)y;
  }
}

// ---------------------------------------------------------------------------
// fp32 -> f16 conversion (weights / activations) and state init.
// ---------------------------------------------------------------------------
__global__ void convert_f32_f16_kernel(const float* __restrict__ s,
                                       _Float16* __restrict__ d, int n) {
  int i = blockIdx.x * blockDim.x + threadIdx.x;
  if (i < n) d[i] = (_Float16)s[i];
}

__global__ void init_state_kernel(const float* __restrict__ h0,
                                  _Float16* __restrict__ hbf,
                                  const float* __restrict__ c0,
                                  float* __restrict__ c, int n) {
  int i = blockIdx.x * blockDim.x + threadIdx.x;
  if (i < n) {
    hbf[i] = (_Float16)h0[i];
    c[i] = c0[i];
  }
}

// ---------------------------------------------------------------------------
// Host driver.
// ---------------------------------------------------------------------------
extern "C" void kernel_launch(void* const* d_in, const int* in_sizes, int n_in,
                              void* d_out, int out_size, void* d_ws, size_t ws_size,
                              hipStream_t stream) {
  const float* x        = (const float*)d_in[0];
  const float* h0       = (const float*)d_in[1];
  const float* c0       = (const float*)d_in[2];
  const float* enc_Wih0 = (const float*)d_in[3];
  const float* enc_Whh0 = (const float*)d_in[4];
  const float* enc_b0   = (const float*)d_in[5];
  const float* enc_Wih1 = (const float*)d_in[6];
  const float* enc_Whh1 = (const float*)d_in[7];
  const float* enc_b1   = (const float*)d_in[8];
  const float* dec_Wih0 = (const float*)d_in[9];
  const float* dec_Whh0 = (const float*)d_in[10];
  const float* dec_b0   = (const float*)d_in[11];
  const float* dec_Wih1 = (const float*)d_in[12];
  const float* dec_Whh1 = (const float*)d_in[13];
  const float* dec_b1   = (const float*)d_in[14];
  const float* lin_W    = (const float*)d_in[15];
  const float* lin_b    = (const float*)d_in[16];
  float* out = (float*)d_out;

  // Carve workspace (256B aligned slices).
  char* wp = (char*)d_ws;
  auto carve_h = [&wp](size_t elems) -> _Float16* {
    _Float16* p = (_Float16*)wp;
    wp += (elems * sizeof(_Float16) + 255) & ~(size_t)255;
    return p;
  };
  auto carve_f = [&wp](size_t elems) -> float* {
    float* p = (float*)wp;
    wp += (elems * sizeof(float) + 255) & ~(size_t)255;
    return p;
  };

  _Float16* x_h    = carve_h((size_t)BB * TT * FF);
  _Float16* w_eih0 = carve_h((size_t)4 * HH * FF);
  _Float16* w_ehh0 = carve_h((size_t)4 * HH * HH);
  _Float16* w_eih1 = carve_h((size_t)4 * HH * HH);
  _Float16* w_ehh1 = carve_h((size_t)4 * HH * HH);
  _Float16* w_dih0 = carve_h((size_t)4 * HH * FF);
  _Float16* w_dhh0 = carve_h((size_t)4 * HH * HH);
  _Float16* w_dih1 = carve_h((size_t)4 * HH * HH);
  _Float16* w_dhh1 = carve_h((size_t)4 * HH * HH);
  _Float16* w_lin  = carve_h((size_t)FF * HH);
  _Float16* hp0[2] = { carve_h((size_t)BB * HH), carve_h((size_t)BB * HH) };
  _Float16* hp1[2] = { carve_h((size_t)BB * HH), carve_h((size_t)BB * HH) };
  _Float16* xd     = carve_h((size_t)BB * FF);
  float* c_l0 = carve_f((size_t)BB * HH);
  float* c_l1 = carve_f((size_t)BB * HH);

  // One-time conversions (re-done each call for determinism).
  {
    int n;
    n = BB * TT * FF;
    convert_f32_f16_kernel<<<(n + 255) / 256, 256, 0, stream>>>(x, x_h, n);
    n = 4 * HH * FF;
    convert_f32_f16_kernel<<<(n + 255) / 256, 256, 0, stream>>>(enc_Wih0, w_eih0, n);
    convert_f32_f16_kernel<<<(n + 255) / 256, 256, 0, stream>>>(dec_Wih0, w_dih0, n);
    n = 4 * HH * HH;
    convert_f32_f16_kernel<<<(n + 255) / 256, 256, 0, stream>>>(enc_Whh0, w_ehh0, n);
    convert_f32_f16_kernel<<<(n + 255) / 256, 256, 0, stream>>>(enc_Wih1, w_eih1, n);
    convert_f32_f16_kernel<<<(n + 255) / 256, 256, 0, stream>>>(enc_Whh1, w_ehh1, n);
    convert_f32_f16_kernel<<<(n + 255) / 256, 256, 0, stream>>>(dec_Whh0, w_dhh0, n);
    convert_f32_f16_kernel<<<(n + 255) / 256, 256, 0, stream>>>(dec_Wih1, w_dih1, n);
    convert_f32_f16_kernel<<<(n + 255) / 256, 256, 0, stream>>>(dec_Whh1, w_dhh1, n);
    n = FF * HH;
    convert_f32_f16_kernel<<<(n + 255) / 256, 256, 0, stream>>>(lin_W, w_lin, n);
    n = BB * HH;
    init_state_kernel<<<(n + 255) / 256, 256, 0, stream>>>(h0, hp0[0], c0, c_l0, n);
    init_state_kernel<<<(n + 255) / 256, 256, 0, stream>>>(h0 + (size_t)BB * HH, hp1[0],
                                                           c0 + (size_t)BB * HH, c_l1, n);
  }

  const dim3 cell_blk(64, 1, 1);
  const dim3 lin_grid(FF / 16, BB / 32, 1);
  const dim3 lin_blk(64, 1, 1);

  // --- Encoder: software-pipelined. layer1(t-1) and layer0(t) depend only on
  // layer0(t-1) output, so they share one launch (gridDim.z = 2).
  // h-state of layer L before step t lives in hpL[t&1]; step t writes hpL[(t+1)&1].
  auto enc_l0 = [&](int t) -> CellArgs {
    CellArgs a;
    a.xin = x_h + (size_t)t * FF; a.ldx = TT * FF; a.Kx = FF;
    a.Wih = w_eih0; a.hin = hp0[t & 1]; a.Whh = w_ehh0;
    a.bias = enc_b0; a.c = c_l0; a.hout = hp0[(t + 1) & 1];
    return a;
  };
  auto enc_l1 = [&](int t) -> CellArgs {
    CellArgs a;
    a.xin = hp0[(t + 1) & 1]; a.ldx = HH; a.Kx = HH;   // layer0 output of step t
    a.Wih = w_eih1; a.hin = hp1[t & 1]; a.Whh = w_ehh1;
    a.bias = enc_b1; a.c = c_l1; a.hout = hp1[(t + 1) & 1];
    return a;
  };

  {
    CellArgs a0 = enc_l0(0);
    lstm_cell_kernel<<<dim3(HH / 16, BB / 32, 1), cell_blk, 0, stream>>>(a0, a0);
  }
  for (int t = 1; t < TT; ++t) {
    CellArgs a0 = enc_l1(t - 1);
    CellArgs a1 = enc_l0(t);
    lstm_cell_kernel<<<dim3(HH / 16, BB / 32, 2), cell_blk, 0, stream>>>(a0, a1);
  }
  {
    CellArgs a0 = enc_l1(TT - 1);
    lstm_cell_kernel<<<dim3(HH / 16, BB / 32, 1), cell_blk, 0, stream>>>(a0, a0);
  }

  // --- Decoder: strictly serial chain (y feeds back as next input).
  // Encoder-final states live at index TT&1 == 0.
  _Float16 *h0c = hp0[0], *h0n = hp0[1], *h1c = hp1[0], *h1n = hp1[1];
  const _Float16* xd_src = x_h + (size_t)(TT - 1) * FF;
  int ldxd = TT * FF;
  for (int t = 0; t < TT; ++t) {
    CellArgs a;
    a.xin = xd_src; a.ldx = ldxd; a.Kx = FF;
    a.Wih = w_dih0; a.hin = h0c; a.Whh = w_dhh0;
    a.bias = dec_b0; a.c = c_l0; a.hout = h0n;
    lstm_cell_kernel<<<dim3(HH / 16, BB / 32, 1), cell_blk, 0, stream>>>(a, a);

    CellArgs b;
    b.xin = h0n; b.ldx = HH; b.Kx = HH;
    b.Wih = w_dih1; b.hin = h1c; b.Whh = w_dhh1;
    b.bias = dec_b1; b.c = c_l1; b.hout = h1n;
    lstm_cell_kernel<<<dim3(HH / 16, BB / 32, 1), cell_blk, 0, stream>>>(b, b);

    linear_out_kernel<<<lin_grid, lin_blk, 0, stream>>>(
        h1n, w_lin, lin_b, out, xd, TT - 1 - t);

    xd_src = xd;
    ldxd = FF;
    { _Float16* tmp = h0c; h0c = h0n; h0n = tmp; }
    { _Float16* tmp = h1c; h1c = h1n; h1n = tmp; }
  }
}
